// Head_78374563217783
// MI455X (gfx1250) — matrix operand order repaired and checked by reading, not compiled, not run on hardware
//
#include <hip/hip_runtime.h>

// ---- problem constants ----
#define SEQ    110
#define HALF   55
#define SEQ_P  112      // 7 * 16
#define NE     102
#define NE_P   128      // 4 * 32 (K padding for f16 WMMA)
#define HEAD   192      // 12 * 16, 6 * 32
#define NWAVE  8

typedef __attribute__((ext_vector_type(16))) _Float16 v16h;
typedef __attribute__((ext_vector_type(8)))  float    v8f;
typedef __attribute__((ext_vector_type(4)))  unsigned int u32x4;
typedef __attribute__((ext_vector_type(8)))  unsigned int u32x8;

union FragU {
    v16h  v;
    uint4 q[2];
};

// A fragment (16x32 f16): lane L -> row M = L&15; K halves {kb..kb+7} in v0-3,
// {kb+16..kb+23} in v4-7, kb = (L>=16)?8:0.  Source is row-major, stride ld halves.
__device__ __forceinline__ v16h load_fragA(const _Float16* base, int row0, int col0,
                                           int ld, int lane) {
    int m  = lane & 15;
    int kb = (lane >> 4) << 3;                  // 0 or 8
    const _Float16* p = base + (row0 + m) * ld + col0 + kb;
    FragU f;
    f.q[0] = *reinterpret_cast<const uint4*>(p);
    f.q[1] = *reinterpret_cast<const uint4*>(p + 16);
    return f.v;
}

// B fragment (32x16 f16) loaded from Bt (row n of Bt = column n of B), row-major,
// stride ld halves: lane L -> col N = L&15; K halves {kb..kb+7},{kb+16..kb+23}, kb = (L>=16)?8:0.
__device__ __forceinline__ v16h load_fragB(const _Float16* bt, int n0, int col0,
                                           int ld, int lane) {
    int n  = lane & 15;
    int kb = (lane >> 4) << 3;                  // 0 or 8
    const _Float16* p = bt + (n0 + n) * ld + col0 + kb;
    FragU f;
    f.q[0] = *reinterpret_cast<const uint4*>(p);
    f.q[1] = *reinterpret_cast<const uint4*>(p + 16);
    return f.v;
}

// ---- LDS layout (bytes) ----
#define XS_OFF 0                                  // x f16 [112][128], reused as P
#define XS_B   (SEQ_P * NE_P * 2)                 // 28672
#define QS_OFF (XS_OFF + XS_B)                    // q f16 [112][192]
#define QS_B   (SEQ_P * HEAD * 2)                 // 43008
#define KS_OFF (QS_OFF + QS_B)                    // k f16 [112][192]
#define KS_B   (SEQ_P * HEAD * 2)                 // 43008
#define VT_OFF (KS_OFF + KS_B)                    // v^T f16 [192][128]
#define VT_B   (HEAD * NE_P * 2)                  // 49152
#define SS_OFF (VT_OFF + VT_B)                    // scores f32 [112][112]; also TDM stage
#define SS_B   (SEQ_P * SEQ_P * 4)                // 50176 (>= 110*102*4 = 44880 stage)
#define SMEM_BYTES (SS_OFF + SS_B)                // 214016

// Transpose + f16-convert the three weight matrices: wt[mat][n][c] = W[c][n]
__global__ void wprep_kernel(const float* __restrict__ Wq, const float* __restrict__ Wk,
                             const float* __restrict__ Wv, _Float16* __restrict__ wt) {
    int i = blockIdx.x * blockDim.x + threadIdx.x;
    if (i >= 3 * HEAD * NE_P) return;
    int mat = i / (HEAD * NE_P);
    int rem = i - mat * (HEAD * NE_P);
    int n   = rem / NE_P;
    int c   = rem - n * NE_P;
    const float* W = (mat == 0) ? Wq : (mat == 1) ? Wk : Wv;
    wt[i] = (c < NE) ? (_Float16)W[c * HEAD + n] : (_Float16)0.0f;
}

__global__ __launch_bounds__(256) void attn_kernel(const float* __restrict__ x,
                                                   const _Float16* __restrict__ wt,
                                                   float* __restrict__ out) {
    extern __shared__ char smem[];
    _Float16* xs = (_Float16*)(smem + XS_OFF);
    _Float16* qs = (_Float16*)(smem + QS_OFF);
    _Float16* ks = (_Float16*)(smem + KS_OFF);
    _Float16* vT = (_Float16*)(smem + VT_OFF);
    float*    ss = (float*)   (smem + SS_OFF);
    float*    stage = ss;   // TDM staging for x[b] f32 (dead until scores phase)

    const int tid  = threadIdx.x;
    const int lane = tid & 31;
    const int wid  = tid >> 5;
    const int b    = blockIdx.x;
    const float* xb = x + (size_t)b * SEQ * NE;

    // ---- wave 0: TDM DMA of x[b] (110x102 f32 tile) global -> LDS stage ----
    // D# per CDNA5 ISA (2D tensor: groups 2/3 = NULL).
    if (wid == 0) {
        unsigned long long ga = (unsigned long long)(uintptr_t)xb;   // 57-bit global addr
        u32x4 g0;
        g0.x = 1u;                                        // count=1, user descriptor
        g0.y = (unsigned int)(uintptr_t)stage;            // lds_addr (bytes)
        g0.z = (unsigned int)ga;                          // global_addr[31:0]
        g0.w = (unsigned int)(ga >> 32) | 0x80000000u;    // global_addr[56:32] | type=2
        u32x8 g1;
        g1[0] = 0x00020000u;                              // data_size=2 (4B); no pad/iter/mask
        g1[1] = ((unsigned int)NE) << 16;                 // tensor_dim0[15:0] @ bits63:48
        g1[2] = ((unsigned int)SEQ) << 16;                // tensor_dim1[15:0] @ bits95:80
        g1[3] = ((unsigned int)NE) << 16;                 // tile_dim0 @ bits127:112
        g1[4] = (unsigned int)SEQ;                        // tile_dim1 @ bits143:128
        g1[5] = (unsigned int)NE;                         // tensor_dim0_stride[31:0]
        g1[6] = 0u;                                       // stride hi / dim1_stride
        g1[7] = 0u;
        asm volatile("tensor_load_to_lds %0, %1" :: "s"(g0), "s"(g1) : "memory");
    }

    // ---- zero padded regions (xs fully, vT fully: avoid NaN garbage in WMMA) ----
    {
        uint4 z = {0u, 0u, 0u, 0u};
        uint4* p0 = (uint4*)xs;
        for (int i = tid; i < XS_B / 16; i += 256) p0[i] = z;
        uint4* p1 = (uint4*)vT;
        for (int i = tid; i < VT_B / 16; i += 256) p1[i] = z;
    }
    if (wid == 0) __builtin_amdgcn_s_wait_tensorcnt(0);   // TDM data landed in LDS
    __syncthreads();

    // ---- convert staged x[b] (f32) -> xs (f16, padded) ----
    for (int i = tid; i < SEQ * NE; i += 256) {
        int s = i / NE;
        int c = i - s * NE;
        xs[s * NE_P + c] = (_Float16)stage[i];
    }
    __syncthreads();

    const int n_c  = lane & 15;           // C/D: lane -> column
    const int mb_c = (lane >> 4) << 3;    // C/D: lanes 16-31 -> rows +8

    // ---- QKV: 3 mats * 7 M-tiles * 12 N-tiles, K = 128 (4 steps) ----
    for (int t = wid; t < 3 * 7 * 12; t += NWAVE) {
        int mat = t / 84;
        int rem = t - mat * 84;
        int tm  = rem / 12;
        int tn  = rem - tm * 12;
        const _Float16* wmat = wt + mat * HEAD * NE_P;
        v8f acc = {};
        #pragma unroll
        for (int kk = 0; kk < 4; ++kk) {
            v16h a  = load_fragA(xs,   tm * 16, kk * 32, NE_P, lane);
            v16h bf = load_fragB(wmat, tn * 16, kk * 32, NE_P, lane);
            acc = __builtin_amdgcn_wmma_f32_16x16x32_f16(false, a, false, bf,
                                                         (short)0, acc, false, false);
        }
        if (mat == 0) {
            #pragma unroll
            for (int r = 0; r < 8; ++r)
                qs[(tm * 16 + mb_c + r) * HEAD + tn * 16 + n_c] = (_Float16)acc[r];
        } else if (mat == 1) {
            #pragma unroll
            for (int r = 0; r < 8; ++r)
                ks[(tm * 16 + mb_c + r) * HEAD + tn * 16 + n_c] = (_Float16)acc[r];
        } else {  // store V transposed: vT[h][t]
            #pragma unroll
            for (int r = 0; r < 8; ++r)
                vT[(tn * 16 + n_c) * NE_P + tm * 16 + mb_c + r] = (_Float16)acc[r];
        }
    }
    __syncthreads();

    // ---- scores = q @ k^T * scale : 7x7 tiles, K = 192 (6 steps) ----
    const float scale = 0.07216878364870323f;   // 192^-0.5
    for (int t = wid; t < 7 * 7; t += NWAVE) {
        int tm = t / 7;
        int tn = t - tm * 7;
        v8f acc = {};
        #pragma unroll
        for (int kk = 0; kk < 6; ++kk) {
            v16h a  = load_fragA(qs, tm * 16, kk * 32, HEAD, lane);
            v16h bf = load_fragB(ks, tn * 16, kk * 32, HEAD, lane);  // row t of k = col t of k^T
            acc = __builtin_amdgcn_wmma_f32_16x16x32_f16(false, a, false, bf,
                                                         (short)0, acc, false, false);
        }
        #pragma unroll
        for (int r = 0; r < 8; ++r)
            ss[(tm * 16 + mb_c + r) * SEQ_P + tn * 16 + n_c] = acc[r] * scale;
    }
    __syncthreads();

    // ---- masked softmax (cross-half only), write P (f16 [112][128], reuses xs) ----
    _Float16* P = xs;
    if (tid < SEQ_P) {
        int  s     = tid;
        bool valid = (s < SEQ);
        int  lo    = (s < HALF) ? HALF : 0;
        int  hi    = (s < HALF) ? SEQ  : HALF;
        float mx = -1e30f;
        if (valid)
            for (int t2 = lo; t2 < hi; ++t2) mx = fmaxf(mx, ss[s * SEQ_P + t2]);
        float sum = 0.0f;
        if (valid)
            for (int t2 = lo; t2 < hi; ++t2) sum += __expf(ss[s * SEQ_P + t2] - mx);
        float inv = (valid && sum > 0.0f) ? (1.0f / sum) : 0.0f;
        for (int t2 = 0; t2 < NE_P; ++t2) {
            float pv = 0.0f;
            if (valid && t2 >= lo && t2 < hi)
                pv = __expf(ss[s * SEQ_P + t2] - mx) * inv;
            P[s * NE_P + t2] = (_Float16)pv;
        }
    }
    __syncthreads();

    // ---- out = P @ V : 7x12 tiles, K = 128 (4 steps; t>=110 columns are zero) ----
    float* outb = out + (size_t)b * SEQ * HEAD;
    for (int t = wid; t < 7 * 12; t += NWAVE) {
        int tm = t / 12;
        int tn = t - tm * 12;
        v8f acc = {};
        #pragma unroll
        for (int kk = 0; kk < 4; ++kk) {
            v16h a  = load_fragA(P,  tm * 16, kk * 32, NE_P, lane);
            v16h bf = load_fragB(vT, tn * 16, kk * 32, NE_P, lane);  // row h of vT = col h of V
            acc = __builtin_amdgcn_wmma_f32_16x16x32_f16(false, a, false, bf,
                                                         (short)0, acc, false, false);
        }
        #pragma unroll
        for (int r = 0; r < 8; ++r) {
            int s = tm * 16 + mb_c + r;
            if (s < SEQ)
                outb[s * HEAD + tn * 16 + n_c] = acc[r];
        }
    }
}

extern "C" void kernel_launch(void* const* d_in, const int* in_sizes, int n_in,
                              void* d_out, int out_size, void* d_ws, size_t ws_size,
                              hipStream_t stream) {
    const float* x  = (const float*)d_in[0];
    const float* Wq = (const float*)d_in[1];
    const float* Wk = (const float*)d_in[2];
    const float* Wv = (const float*)d_in[3];
    float*       out = (float*)d_out;
    _Float16*    wt  = (_Float16*)d_ws;     // 3*192*128*2 = 147456 bytes

    int B = in_sizes[0] / (SEQ * NE);       // 8192

    int wtot = 3 * HEAD * NE_P;
    wprep_kernel<<<(wtot + 255) / 256, 256, 0, stream>>>(Wq, Wk, Wv, wt);
    attn_kernel<<<B, 256, SMEM_BYTES, stream>>>(x, wt, out);
}
